// MultiheadSelectiveAttention_21354577395912
// MI455X (gfx1250) — compile-verified, matrix operand order checked by reading, not run on hardware
//
#include <hip/hip_runtime.h>
#include <hip/hip_bf16.h>

typedef __attribute__((ext_vector_type(16))) __bf16 v16bf;
typedef __attribute__((ext_vector_type(8)))  __bf16 v8bf;
typedef __attribute__((ext_vector_type(8)))  float  v8f;

#define WMMA_BF16(a, b, c) \
  __builtin_amdgcn_wmma_f32_16x16x32_bf16(false, (a), false, (b), (short)0, (c), false, false)

// Load a 16-bit A/B fragment: lane holds K = klo..klo+7 (elems 0..7) and
// K = klo+16..klo+23 (elems 8..15); both halves are 16B-contiguous.
static __device__ __forceinline__ v16bf load_frag(const __bf16* __restrict__ p) {
  v8bf lo = *(const v8bf*)(p);
  v8bf hi = *(const v8bf*)(p + 16);
  v16bf r;
#pragma unroll
  for (int i = 0; i < 8; ++i) { r[i] = lo[i]; r[8 + i] = hi[i]; }
  return r;
}

// CDNA5 async DMA: global -> LDS, 16 bytes per lane, tracked by ASYNCcnt.
static __device__ __forceinline__ void async_stage_b128(unsigned lds_off, const __bf16* gptr) {
  unsigned long long ga = (unsigned long long)(uintptr_t)gptr;
  asm volatile("global_load_async_to_lds_b128 %0, %1, off"
               :: "v"(lds_off), "v"(ga) : "memory");
}
static __device__ __forceinline__ void wait_asynccnt0() {
  asm volatile("s_wait_asynccnt 0x0" ::: "memory");
}

// ---------------- elementwise f32 -> bf16 convert ----------------
__global__ void cvt_f32_bf16(const float* __restrict__ in, __bf16* __restrict__ out, int n) {
  int i = blockIdx.x * blockDim.x + threadIdx.x;
  if (i < n) out[i] = (__bf16)in[i];
}

// ---------------- tiled transpose + convert: (B,R,C) f32 -> (B,C,R) bf16 ----------------
__global__ __launch_bounds__(256) void transpose_cvt(const float* __restrict__ in,
                                                     __bf16* __restrict__ out,
                                                     int R, int C) {
  __shared__ float tile[32][33];
  const int b = blockIdx.z;
  const int r0 = blockIdx.x * 32, c0 = blockIdx.y * 32;
  const int tx = threadIdx.x & 31, ty = threadIdx.x >> 5;  // 32 x 8
  const float* src = in + ((size_t)b * R + r0) * C + c0;
#pragma unroll
  for (int i = ty; i < 32; i += 8) tile[i][tx] = src[(size_t)i * C + tx];
  __syncthreads();
  __bf16* dst = out + ((size_t)b * C + c0) * R + r0;
#pragma unroll
  for (int i = ty; i < 32; i += 8) dst[(size_t)i * R + tx] = (__bf16)tile[tx][i];
}

// ---------------- GEMM: C[M,N] = A[M,K] @ B[N,K]^T (bf16 in, f32 out) ----------------
// 256 threads = 8 waves; wave tile 32x64; block tile 256x64.
// B-tile (64x32 bf16 = 4KB) cooperatively staged into LDS with async DMA, double-buffered.
__global__ __launch_bounds__(256) void gemm_bf16_f32(const __bf16* __restrict__ A,
                                                     const __bf16* __restrict__ Bw,
                                                     float* __restrict__ C,
                                                     int M, int N, int K) {
  __shared__ __align__(16) __bf16 Bt[2][64][32];
  const int tid = threadIdx.x;
  const int lane = tid & 31, wid = tid >> 5;
  const int hl = lane >> 4, l16 = lane & 15;
  const int mbase = blockIdx.y * 256 + wid * 32;
  const int nbase = blockIdx.x * 64;

  // per-thread staging slot: 16B of the 4KB tile
  const int srow = tid >> 2;                 // 0..63
  const int scol = (tid & 3) * 8;            // 0,8,16,24
  const __bf16* sgp = Bw + (size_t)(nbase + srow) * K + scol;
  const unsigned slds0 = (unsigned)(uintptr_t)&Bt[0][srow][scol];
  const unsigned slds1 = (unsigned)(uintptr_t)&Bt[1][srow][scol];

  v8f acc[2][4] = {};
  const __bf16* arow0 = A + (size_t)(mbase + l16) * K;
  const __bf16* arow1 = A + (size_t)(mbase + 16 + l16) * K;

  async_stage_b128(slds0, sgp);              // prime k0 = 0
  int buf = 0;
  for (int k0 = 0; k0 < K; k0 += 32) {
    wait_asynccnt0();
    __syncthreads();                         // current buffer ready; previous reads done
    if (k0 + 32 < K)
      async_stage_b128(buf ? slds0 : slds1, sgp + k0 + 32);
    const int klo = k0 + hl * 8;
    v16bf a0 = load_frag(arow0 + klo);
    v16bf a1 = load_frag(arow1 + klo);
#pragma unroll
    for (int j = 0; j < 4; ++j) {
      v16bf b = load_frag(&Bt[buf][j * 16 + l16][hl * 8]);
      acc[0][j] = WMMA_BF16(a0, b, acc[0][j]);
      acc[1][j] = WMMA_BF16(a1, b, acc[1][j]);
    }
    buf ^= 1;
  }
#pragma unroll
  for (int s = 0; s < 2; ++s)
#pragma unroll
    for (int j = 0; j < 4; ++j)
#pragma unroll
      for (int r = 0; r < 8; ++r) {
        const int row = mbase + s * 16 + hl * 8 + r;
        const int col = nbase + j * 16 + l16;
        C[(size_t)row * N + col] = acc[s][j][r];
      }
}

// ---------------- row LayerNorm: f32 in -> bf16 out ----------------
__global__ __launch_bounds__(256) void ln_rows(const float* __restrict__ Y,
                                               const float* __restrict__ g,
                                               const float* __restrict__ bb,
                                               __bf16* __restrict__ out, int D) {
  const int row = blockIdx.x;
  const float* y = Y + (size_t)row * D;
  float s = 0.f, s2 = 0.f;
  for (int i = threadIdx.x; i < D; i += blockDim.x) { float v = y[i]; s += v; s2 += v * v; }
#pragma unroll
  for (int o = 16; o > 0; o >>= 1) { s += __shfl_xor(s, o, 32); s2 += __shfl_xor(s2, o, 32); }
  __shared__ float ps[8], ps2[8];
  const int wid = threadIdx.x >> 5, lane = threadIdx.x & 31;
  if (lane == 0) { ps[wid] = s; ps2[wid] = s2; }
  __syncthreads();
  if (wid == 0) {
    s  = (lane < 8) ? ps[lane]  : 0.f;
    s2 = (lane < 8) ? ps2[lane] : 0.f;
#pragma unroll
    for (int o = 4; o > 0; o >>= 1) { s += __shfl_xor(s, o, 32); s2 += __shfl_xor(s2, o, 32); }
    if (lane == 0) { ps[0] = s; ps2[0] = s2; }
  }
  __syncthreads();
  const float mu  = ps[0] / (float)D;
  const float var = ps2[0] / (float)D - mu * mu;
  const float inv = rsqrtf(var + 1e-5f);
  for (int i = threadIdx.x; i < D; i += blockDim.x)
    out[(size_t)row * D + i] = (__bf16)(((y[i] - mu) * inv) * g[i] + bb[i]);
}

// ---------------- head-0 selective gate S (pre-cumsum) ----------------
// S[b,n,m] = (m>n || m==n || m==0) ? 0 : max(0, (Q0[n]·K0[m])/8)
__global__ __launch_bounds__(128) void selective_S_kernel(const __bf16* __restrict__ Qb,
                                                          const __bf16* __restrict__ Kb,
                                                          float* __restrict__ S,
                                                          int Nn, int D, float scale) {
  const int lane = threadIdx.x & 31, wid = threadIdx.x >> 5;
  const int hl = lane >> 4, l16 = lane & 15;
  const int b = blockIdx.z;
  const int nbase = blockIdx.y * 16;
  const int mbase = (blockIdx.x * 4 + wid) * 16;
  float* Sout = S + (size_t)b * Nn * Nn;

  if (mbase > nbase + 15) {   // fully masked tile: must still write zeros (ws is poisoned)
#pragma unroll
    for (int r = 0; r < 8; ++r)
      Sout[(size_t)(nbase + hl * 8 + r) * Nn + (mbase + l16)] = 0.f;
    return;
  }
  const __bf16* qrow = Qb + ((size_t)b * Nn + nbase + l16) * D;   // head 0 at offset 0
  const __bf16* krow = Kb + ((size_t)b * Nn + mbase + l16) * D;
  v8f acc = {};
#pragma unroll
  for (int ks = 0; ks < 2; ++ks) {
    const int klo = ks * 32 + hl * 8;
    v16bf a = load_frag(qrow + klo);
    v16bf k = load_frag(krow + klo);
    acc = WMMA_BF16(a, k, acc);
  }
#pragma unroll
  for (int r = 0; r < 8; ++r) {
    const int row = nbase + hl * 8 + r;
    const int col = mbase + l16;
    float v = acc[r] * scale;
    v = (col > row || col == row || col == 0) ? 0.f : fmaxf(v, 0.f);
    Sout[(size_t)row * Nn + col] = v;
  }
}

// ---------------- exclusive cumsum along n (in place) ----------------
__global__ void excl_cumsum_kernel(float* __restrict__ S, int Bn, int Nn) {
  const int t = blockIdx.x * blockDim.x + threadIdx.x;
  if (t >= Bn * Nn) return;
  const int b = t / Nn, m = t % Nn;
  float* p = S + (size_t)b * Nn * Nn + m;
  float run = 0.f;
  for (int n = 0; n < Nn; n += 4) {           // unroll: independent loads batched
    float t0 = p[0], t1 = p[(size_t)Nn], t2 = p[(size_t)2 * Nn], t3 = p[(size_t)3 * Nn];
    p[0] = run; run += t0;
    p[(size_t)Nn] = run; run += t1;
    p[(size_t)2 * Nn] = run; run += t2;
    p[(size_t)3 * Nn] = run; run += t3;
    p += (size_t)4 * Nn;
  }
}

// ---------------- flash attention with F-mask subtraction ----------------
// one wave = 32 query rows (2 subtiles) x one (b,h); m-blocks of 32 keys.
// K B-frags shared by 2 Q A-frags; V B-frags shared by 2 P A-frags. V is (B,D,S).
__global__ __launch_bounds__(128) void attn_kernel(const __bf16* __restrict__ Qb,
                                                   const __bf16* __restrict__ Kb,
                                                   const __bf16* __restrict__ Vt,
                                                   const float* __restrict__ F,
                                                   __bf16* __restrict__ Ob,
                                                   int Nn, int dh, int D, float scale) {
  __shared__ __align__(16) __bf16 Plds[4][32][32];
  const int lane = threadIdx.x & 31, wid = threadIdx.x >> 5;
  const int hl = lane >> 4, l16 = lane & 15;
  const int gw = blockIdx.x * 4 + wid;          // 2048 waves total
  const int qtile = gw & 63;                    // Nn/32 = 64
  const int h = (gw >> 6) & 15;
  const int b = gw >> 10;
  const int nbase = qtile * 32;
  const size_t hoff = (size_t)h * dh;

  // Q A-frags: 2 q-subtiles x 2 k-steps
  v16bf aQ[2][2];
#pragma unroll
  for (int s = 0; s < 2; ++s) {
    const __bf16* qrow = Qb + ((size_t)b * Nn + nbase + s * 16 + l16) * D + hoff;
    aQ[s][0] = load_frag(qrow + hl * 8);
    aQ[s][1] = load_frag(qrow + 32 + hl * 8);
  }
  const __bf16* vbase = Vt + ((size_t)b * D + hoff) * Nn;

  v8f oacc[2][4] = {};
  float mrow[2][8], lrow[2][8];
#pragma unroll
  for (int s = 0; s < 2; ++s)
#pragma unroll
    for (int r = 0; r < 8; ++r) { mrow[s][r] = -1e30f; lrow[s][r] = 0.f; }

  const float* Fb = F + (size_t)b * Nn * Nn;
  const int mbmax = qtile + 1;                  // causal: keys <= nbase+31

  for (int mb = 0; mb < mbmax; ++mb) {
    const int mstart = mb * 32;
    // ---- QK^T: 32 x 32 logits (4 C-tiles); K frags reused by both q-subtiles ----
    v8f cs[2][2] = {};
#pragma unroll
    for (int ks = 0; ks < 2; ++ks) {
      const int klo = ks * 32 + hl * 8;
      const __bf16* k0p = Kb + ((size_t)b * Nn + mstart + l16) * D + hoff + klo;
      v16bf bk0 = load_frag(k0p);
      v16bf bk1 = load_frag(k0p + (size_t)16 * D);
#pragma unroll
      for (int s = 0; s < 2; ++s) {
        cs[s][0] = WMMA_BF16(aQ[s][ks], bk0, cs[s][0]);
        cs[s][1] = WMMA_BF16(aQ[s][ks], bk1, cs[s][1]);
      }
    }
    // ---- per q-subtile: mask, F subtraction, online softmax, stage P ----
#pragma unroll
    for (int s = 0; s < 2; ++s) {
      float p0[8], p1[8], t[8], sr[8];
#pragma unroll
      for (int r = 0; r < 8; ++r) {
        const int row = nbase + s * 16 + hl * 8 + r;
        const int cA = mstart + l16, cB = mstart + 16 + l16;
        p0[r] = (cA > row) ? -1e30f : cs[s][0][r] * scale - Fb[(size_t)row * Nn + cA];
        p1[r] = (cB > row) ? -1e30f : cs[s][1][r] * scale - Fb[(size_t)row * Nn + cB];
        t[r] = fmaxf(p0[r], p1[r]);
      }
#pragma unroll
      for (int o = 1; o < 16; o <<= 1)
#pragma unroll
        for (int r = 0; r < 8; ++r) t[r] = fmaxf(t[r], __shfl_xor(t[r], o, 32));
#pragma unroll
      for (int r = 0; r < 8; ++r) {
        const float mn = fmaxf(mrow[s][r], t[r]);
        const float sc = __expf(mrow[s][r] - mn);
        mrow[s][r] = mn;
        p0[r] = __expf(p0[r] - mn);
        p1[r] = __expf(p1[r] - mn);
        sr[r] = p0[r] + p1[r];
        lrow[s][r] *= sc;
#pragma unroll
        for (int j = 0; j < 4; ++j) oacc[s][j][r] *= sc;
      }
#pragma unroll
      for (int o = 1; o < 16; o <<= 1)
#pragma unroll
        for (int r = 0; r < 8; ++r) sr[r] += __shfl_xor(sr[r], o, 32);
#pragma unroll
      for (int r = 0; r < 8; ++r) lrow[s][r] += sr[r];
#pragma unroll
      for (int r = 0; r < 8; ++r) {
        Plds[wid][s * 16 + hl * 8 + r][l16]      = (__bf16)p0[r];
        Plds[wid][s * 16 + hl * 8 + r][16 + l16] = (__bf16)p1[r];
      }
    }
    // ---- P A-frags from LDS (two ds_load_b128 each) ----
    v16bf aP0 = load_frag(&Plds[wid][l16][hl * 8]);
    v16bf aP1 = load_frag(&Plds[wid][16 + l16][hl * 8]);
    // ---- P @ V: V B-frags reused by both q-subtiles ----
#pragma unroll
    for (int j = 0; j < 4; ++j) {
      v16bf bV = load_frag(vbase + (size_t)(j * 16 + l16) * Nn + mstart + hl * 8);
      oacc[0][j] = WMMA_BF16(aP0, bV, oacc[0][j]);
      oacc[1][j] = WMMA_BF16(aP1, bV, oacc[1][j]);
    }
  }
  // ---- epilogue: normalize, write bf16 O ----
#pragma unroll
  for (int s = 0; s < 2; ++s)
#pragma unroll
    for (int r = 0; r < 8; ++r) {
      const float inv = 1.f / lrow[s][r];
      const int row = nbase + s * 16 + hl * 8 + r;
      __bf16* orow = Ob + ((size_t)b * Nn + row) * D + hoff;
#pragma unroll
      for (int j = 0; j < 4; ++j) orow[j * 16 + l16] = (__bf16)(oacc[s][j][r] * inv);
    }
}

// =======================================================================
extern "C" void kernel_launch(void* const* d_in, const int* in_sizes, int n_in,
                              void* d_out, int out_size, void* d_ws, size_t ws_size,
                              hipStream_t stream) {
  (void)in_sizes; (void)n_in; (void)out_size; (void)ws_size;
  constexpr int B = 2, S = 2048, D = 1024, H = 16, DH = 64;
  constexpr int M4 = B * S;            // 4096 rows
  constexpr float SCALE = 0.125f;      // 1/sqrt(64)

  const float* X  = (const float*)d_in[0];
  const float* Wq = (const float*)d_in[1];
  const float* Wk = (const float*)d_in[2];
  const float* Wv = (const float*)d_in[3];
  const float* Wo = (const float*)d_in[4];
  const float* gq = (const float*)d_in[5];
  const float* bq = (const float*)d_in[6];
  const float* gk = (const float*)d_in[7];
  const float* bk = (const float*)d_in[8];
  float* out = (float*)d_out;

  char* ws = (char*)d_ws;
  constexpr size_t MB = 1024 * 1024;
  __bf16* Xb  = (__bf16*)(ws + 0);        // 8MB  (consumed after projections)
  __bf16* Wqb = (__bf16*)(ws + 8  * MB);
  __bf16* Wkb = (__bf16*)(ws + 10 * MB);
  __bf16* Wvb = (__bf16*)(ws + 12 * MB);
  __bf16* Wob = (__bf16*)(ws + 14 * MB);
  float*  Qraw = (float*)(ws + 16 * MB);  // 16MB each raw f32 GEMM outputs
  float*  Kraw = (float*)(ws + 32 * MB);
  float*  Vraw = (float*)(ws + 48 * MB);
  float*  Sbuf = (float*)(ws + 16 * MB);  // 32MB; aliases Qraw/Kraw after LN consumes them
  __bf16* Qb  = (__bf16*)(ws + 64 * MB);  // 8MB each
  __bf16* Kb  = (__bf16*)(ws + 72 * MB);
  __bf16* Vt  = (__bf16*)(ws + 80 * MB);  // V transposed: (B, D, S) bf16
  __bf16* Ob  = (__bf16*)(ws + 0);        // aliases Xb (consumed) -- total ws use: 88MB

  // 1) convert inputs to bf16
  cvt_f32_bf16<<<(M4 * D + 255) / 256, 256, 0, stream>>>(X, Xb, M4 * D);
  cvt_f32_bf16<<<(D * D + 255) / 256, 256, 0, stream>>>(Wq, Wqb, D * D);
  cvt_f32_bf16<<<(D * D + 255) / 256, 256, 0, stream>>>(Wk, Wkb, D * D);
  cvt_f32_bf16<<<(D * D + 255) / 256, 256, 0, stream>>>(Wv, Wvb, D * D);
  cvt_f32_bf16<<<(D * D + 255) / 256, 256, 0, stream>>>(Wo, Wob, D * D);

  // 2) projections (WMMA GEMM, async-LDS double-buffered B tiles)
  dim3 ggrid(D / 64, M4 / 256);
  gemm_bf16_f32<<<ggrid, 256, 0, stream>>>(Xb, Wqb, Qraw, M4, D, D);
  gemm_bf16_f32<<<ggrid, 256, 0, stream>>>(Xb, Wkb, Kraw, M4, D, D);
  gemm_bf16_f32<<<ggrid, 256, 0, stream>>>(Xb, Wvb, Vraw, M4, D, D);

  // 3) layernorm Q/K -> bf16 ; transpose+convert V -> (B,D,S) bf16
  ln_rows<<<M4, 256, 0, stream>>>(Qraw, gq, bq, Qb, D);
  ln_rows<<<M4, 256, 0, stream>>>(Kraw, gk, bk, Kb, D);
  transpose_cvt<<<dim3(S / 32, D / 32, B), 256, 0, stream>>>(Vraw, Vt, S, D);

  // 4) selective gate: head-0 logits -> S, then exclusive cumsum -> F
  selective_S_kernel<<<dim3(S / 64, S / 16, B), 128, 0, stream>>>(Qb, Kb, Sbuf, S, D, SCALE);
  excl_cumsum_kernel<<<(B * S + 255) / 256, 256, 0, stream>>>(Sbuf, B, S);

  // 5) attention (WMMA flash, 32-query waves) -> Ob bf16
  attn_kernel<<<(B * H * (S / 32)) / 4, 128, 0, stream>>>(Qb, Kb, Vt, Sbuf, Ob, S, DH, D, SCALE);

  // 6) output projection -> d_out f32
  gemm_bf16_f32<<<ggrid, 256, 0, stream>>>(Ob, Wob, out, M4, D, D);
}
